// Hippocampus_90890097918026
// MI455X (gfx1250) — matrix operand order repaired and checked by reading, not compiled, not run on hardware
//
#include <hip/hip_runtime.h>

#define B_Q     2048
#define N_K     65536
#define D_F     256
#define TOPK    32
#define NSPLIT  8
#define SEG     (N_K / NSPLIT)        // 8192 keys per segment
#define MTILE   16                    // query rows per block
#define WAVES   8
#define CHUNK_COLS (WAVES * 16)       // 128 keys per chunk
#define NCHUNK  (SEG / CHUNK_COLS)    // 64 chunks
#define INV_TAU 5.0f
#define WAVE_LDS 16384                // 2 x 8KB staging buffers per wave

typedef __attribute__((ext_vector_type(16))) _Float16 v16h;
typedef __attribute__((ext_vector_type(8)))  _Float16 v8h;
typedef __attribute__((ext_vector_type(4)))  _Float16 v4h;
typedef __attribute__((ext_vector_type(8)))  float    v8f;
typedef __attribute__((ext_vector_type(4)))  float    v4f;

#define AS3 __attribute__((address_space(3)))

// -------- LDS helpers (explicit addrspace(3) -> guaranteed ds_* ops) --------
__device__ __forceinline__ v16h lds_read_v16h(unsigned addr) {
    return *(AS3 const v16h*)addr;
}
__device__ __forceinline__ void lds_write_f32(unsigned addr, float v) {
    *(AS3 float*)addr = v;
}
__device__ __forceinline__ float lds_read_f32(unsigned addr) {
    return *(AS3 const float*)addr;
}
__device__ __forceinline__ void lds_write_i32(unsigned addr, int v) {
    *(AS3 int*)addr = v;
}
__device__ __forceinline__ int lds_read_i32(unsigned addr) {
    return *(AS3 const int*)addr;
}

// 8KB contiguous global -> LDS async copy (one wave, 16 x b128 per lane).
// INST_OFFSET is added to both the LDS and global addresses (ISA 10.x async).
__device__ __forceinline__ void stage8k_async(unsigned lds_dst, unsigned long long gsrc) {
#define STAGE_ONE(OFF)                                                          \
    asm volatile("global_load_async_to_lds_b128 %0, %1, off offset:" #OFF       \
                 :: "v"(lds_dst), "v"(gsrc) : "memory")
    STAGE_ONE(0);    STAGE_ONE(512);  STAGE_ONE(1024); STAGE_ONE(1536);
    STAGE_ONE(2048); STAGE_ONE(2560); STAGE_ONE(3072); STAGE_ONE(3584);
    STAGE_ONE(4096); STAGE_ONE(4608); STAGE_ONE(5120); STAGE_ONE(5632);
    STAGE_ONE(6144); STAGE_ONE(6656); STAGE_ONE(7168); STAGE_ONE(7680);
#undef STAGE_ONE
}
__device__ __forceinline__ void wait_async0() {
    asm volatile("s_wait_asynccnt 0x0" ::: "memory");
}

// -------- wave32 shuffle helpers --------
__device__ __forceinline__ void wave_argmin(float v, int lane, float& mv, int& ml) {
    mv = v; ml = lane;
#pragma unroll
    for (int off = 16; off >= 1; off >>= 1) {
        float ov = __shfl_xor(mv, off, 32);
        int   ol = __shfl_xor(ml, off, 32);
        if (ov < mv || (ov == mv && ol < ml)) { mv = ov; ml = ol; }
    }
}
__device__ __forceinline__ float wave_max(float v) {
#pragma unroll
    for (int off = 16; off >= 1; off >>= 1) v = fmaxf(v, __shfl_xor(v, off, 32));
    return v;
}
__device__ __forceinline__ float wave_sum(float v) {
#pragma unroll
    for (int off = 16; off >= 1; off >>= 1) v += __shfl_xor(v, off, 32);
    return v;
}

// ------------------------- prep kernels -------------------------------
__global__ void init_max_kernel(int* maxslot) {
    if (threadIdx.x == 0) *maxslot = 0;
}

__global__ __launch_bounds__(256) void prep_lastused_kernel(
        const float* __restrict__ lu, float* __restrict__ lastout, int* __restrict__ maxslot) {
    int i = blockIdx.x * blockDim.x + threadIdx.x;
    if (i < N_K) {
        float v = lu[i];
        lastout[i] = v;
        atomicMax(maxslot, __float_as_int(v));   // exact for non-negative LRU counters
    }
}

__global__ __launch_bounds__(256) void q_normalize_kernel(
        const float* __restrict__ q, _Float16* __restrict__ qh) {
    const int lane = threadIdx.x & 31;
    const int w    = threadIdx.x >> 5;
    const int row  = blockIdx.x * 8 + w;
    const v4f* p = (const v4f*)(q + (size_t)row * D_F) + lane * 2;
    v4f x0 = p[0], x1 = p[1];
    float ss = 0.f;
#pragma unroll
    for (int k = 0; k < 4; ++k) ss += x0[k] * x0[k] + x1[k] * x1[k];
    ss = wave_sum(ss);
    float sc = 1.0f / fmaxf(sqrtf(ss), 1e-12f);
    v8h h;
#pragma unroll
    for (int k = 0; k < 4; ++k) {
        h[k]     = (_Float16)(x0[k] * sc);
        h[k + 4] = (_Float16)(x1[k] * sc);
    }
    *(v8h*)(qh + (size_t)row * D_F + lane * 8) = h;
}

__global__ __launch_bounds__(256) void k_convert_kernel(
        const float* __restrict__ K, _Float16* __restrict__ kh) {
    size_t i = ((size_t)blockIdx.x * blockDim.x + threadIdx.x) * 4;
    if (i < (size_t)N_K * D_F) {
        v4f x = *(const v4f*)(K + i);
        v4h h;
#pragma unroll
        for (int k = 0; k < 4; ++k) h[k] = (_Float16)x[k];
        *(v4h*)(kh + i) = h;
    }
}

// --------------- stage 2: WMMA similarity + streaming top-k -----------
// grid = (B_Q/MTILE, NSPLIT), block = 256 (8 waves), dynamic LDS = 128KB
__global__ __launch_bounds__(256) void topk_gemm_kernel(
        const _Float16* __restrict__ qh, const _Float16* __restrict__ kh,
        float* __restrict__ cv_out, int* __restrict__ ci_out) {
    extern __shared__ char smem[];
    const unsigned smbase = (unsigned)(size_t)smem;

    const int lane    = threadIdx.x & 31;
    const int w       = threadIdx.x >> 5;
    const int qbase   = blockIdx.x * MTILE;
    const int seg     = blockIdx.y;
    const int segbase = seg * SEG;

    // Resident A operand: 16x32 f16 tiles (contiguous k order, k = 16h + e; B matches).
    const int arow = lane & 15;
    const int aoff = (lane < 16) ? 0 : 16;
    v16h a[8];
#pragma unroll
    for (int s = 0; s < 8; ++s) {
        const _Float16* p = qh + (size_t)(qbase + arow) * D_F + s * 32 + aoff;
        v8h lo = *(const v8h*)p;
        v8h hi = *(const v8h*)(p + 8);
        a[s] = __builtin_shufflevector(lo, hi, 0,1,2,3,4,5,6,7,8,9,10,11,12,13,14,15);
    }

    // Lane-resident exact top-32 per query row (replace-min streaming).
    float thr[MTILE], tv[MTILE];
    int   ti[MTILE];
#pragma unroll
    for (int r = 0; r < MTILE; ++r) {
        thr[r] = -__builtin_inff();
        tv[r]  = -__builtin_inff();
        ti[r]  = segbase;
    }

    // Per-wave double-buffered async staging of its own 16 keys (8KB/chunk).
    const unsigned ldsBuf0 = smbase + w * WAVE_LDS + lane * 16;
    const unsigned ldsBuf1 = ldsBuf0 + 8192;
    unsigned long long gsrc =
        (unsigned long long)(size_t)(kh + (size_t)(segbase + w * 16) * D_F) + lane * 16;

    stage8k_async(ldsBuf0, gsrc);          // chunk 0 -> buf0
    gsrc += (unsigned long long)CHUNK_COLS * D_F * 2;

    const unsigned bBase0 = smbase + w * WAVE_LDS + (lane & 15) * 512 + ((lane < 16) ? 0 : 32);

    for (int ch = 0; ch < NCHUNK; ++ch) {
        const int colbase = segbase + ch * CHUNK_COLS + w * 16;

        wait_async0();                     // buffer (ch&1) staged a full chunk ago
        if (ch + 1 < NCHUNK) {
            stage8k_async((ch & 1) ? ldsBuf0 : ldsBuf1, gsrc);
            gsrc += (unsigned long long)CHUNK_COLS * D_F * 2;
        }

        // Batched LDS reads of all 8 B slices (single clause), then WMMAs with
        // graduated dscnt waits. sched_barrier(0) keeps the loads from sinking.
        const unsigned bb = bBase0 + ((ch & 1) ? 8192u : 0u);
        v16h b[8];
#pragma unroll
        for (int s = 0; s < 8; ++s) b[s] = lds_read_v16h(bb + s * 64);
        __builtin_amdgcn_sched_barrier(0);

        v8f c = {};
#pragma unroll
        for (int s = 0; s < 8; ++s)
            c = __builtin_amdgcn_wmma_f32_16x16x32_f16(
                    false, a[s], false, b[s], (short)0, c, false, false);

        // C layout: VGPR i -> (M=i, lanes 0-15) and (M=i+8, lanes 16-31).
#pragma unroll
        for (int r = 0; r < MTILE; ++r) {
            const float cv       = c[r & 7];
            const int   halfBase = (r < 8) ? 0 : 16;
            const bool  mine     = ((lane >> 4) == ((r < 8) ? 0 : 1));
            if ((unsigned)__ballot(mine && (cv > thr[r]))) {   // scalar-only gate
#pragma unroll 1
                for (int j = 0; j < 16; ++j) {
                    float v = __shfl(cv, halfBase + j, 32);
                    if (v > thr[r]) {
                        float mv; int ml;
                        wave_argmin(tv[r], lane, mv, ml);
                        if (v > mv && lane == ml) { tv[r] = v; ti[r] = colbase + j; }
                        thr[r] = mv;   // lower bound on current min -> exact gate
                    }
                }
            }
        }
    }

    // Intra-block merge (reuses staging LDS after all async traffic drained).
    wait_async0();
    __syncthreads();
    const unsigned lvBase = smbase;               // 16 x 256 f32
    const unsigned liBase = smbase + 16384;       // 16 x 256 i32
#pragma unroll
    for (int r = 0; r < MTILE; ++r) {
        lds_write_f32(lvBase + (r * 256 + w * 32 + lane) * 4, tv[r]);
        lds_write_i32(liBase + (r * 256 + w * 32 + lane) * 4, ti[r]);
    }
    __syncthreads();

#pragma unroll 1
    for (int rr = 0; rr < 2; ++rr) {
        const int r = w * 2 + rr;
        float bv = lds_read_f32(lvBase + (r * 256 + lane) * 4);
        int   bi = lds_read_i32(liBase + (r * 256 + lane) * 4);
        float mv; int ml;
        wave_argmin(bv, lane, mv, ml);
        float th = mv;
#pragma unroll 1
        for (int j = 1; j < WAVES; ++j) {
            float v  = lds_read_f32(lvBase + (r * 256 + j * 32 + lane) * 4);
            int   id = lds_read_i32(liBase + (r * 256 + j * 32 + lane) * 4);
#pragma unroll 1
            for (int l = 0; l < 32; ++l) {
                float vv = __shfl(v, l, 32);
                if (vv > th) {
                    int ii = __shfl(id, l, 32);
                    wave_argmin(bv, lane, mv, ml);
                    if (vv > mv && lane == ml) { bv = vv; bi = ii; }
                    th = mv;
                }
            }
        }
        size_t o = ((size_t)(qbase + r) * NSPLIT + seg) * TOPK + lane;
        cv_out[o] = bv;
        ci_out[o] = bi;
    }
}

// ---- stage 3: cross-segment merge + softmax + V gather + LRU scatter ----
__global__ __launch_bounds__(256) void merge_softmax_out_kernel(
        const float* __restrict__ cv, const int* __restrict__ ci,
        const float* __restrict__ V, const int* __restrict__ maxslot,
        float* __restrict__ out, float* __restrict__ lastout) {
    const int lane = threadIdx.x & 31;
    const int w    = threadIdx.x >> 5;
    const int q    = blockIdx.x * 8 + w;
    const size_t base = (size_t)q * NSPLIT * TOPK;

    float bv = cv[base + lane];
    int   bi = ci[base + lane];
    float mv; int ml;
    wave_argmin(bv, lane, mv, ml);
    float th = mv;
#pragma unroll 1
    for (int s = 1; s < NSPLIT; ++s) {
        float v  = cv[base + s * TOPK + lane];
        int   id = ci[base + s * TOPK + lane];
#pragma unroll 1
        for (int l = 0; l < 32; ++l) {
            float vv = __shfl(v, l, 32);
            if (vv > th) {
                int ii = __shfl(id, l, 32);
                wave_argmin(bv, lane, mv, ml);
                if (vv > mv && lane == ml) { bv = vv; bi = ii; }
                th = mv;
            }
        }
    }

    // softmax over the 32 selected logits (scaled by 1/tau)
    float mx = wave_max(bv);
    float e  = __expf((bv - mx) * INV_TAU);
    float sm = wave_sum(e);
    float wgt = e / sm;

    // out[q] = sum_j wgt_j * V[idx_j]
    v4f acc0 = {}, acc1 = {};
    const v4f* V4 = (const v4f*)V;
#pragma unroll 1
    for (int j = 0; j < TOPK; ++j) {
        float wj = __shfl(wgt, j, 32);
        int   ij = __shfl(bi, j, 32);
        const v4f* vp = V4 + (size_t)ij * (D_F / 4) + lane * 2;
        acc0 += wj * vp[0];
        acc1 += wj * vp[1];
    }
    v4f* o4 = (v4f*)(out + (size_t)q * D_F) + lane * 2;
    o4[0] = acc0;
    o4[1] = acc1;

    // LRU scatter (same value everywhere -> write races are benign)
    float counter = __int_as_float(*maxslot) + 1.0f;
    lastout[bi] = counter;
}

// ------------------------------ launch --------------------------------
extern "C" void kernel_launch(void* const* d_in, const int* in_sizes, int n_in,
                              void* d_out, int out_size, void* d_ws, size_t ws_size,
                              hipStream_t stream) {
    const float* q_in = (const float*)d_in[0];
    const float* K_in = (const float*)d_in[1];
    const float* V_in = (const float*)d_in[2];
    const float* lu   = (const float*)d_in[3];
    (void)in_sizes; (void)n_in; (void)out_size; (void)ws_size;

    float* out     = (float*)d_out;                       // [B, D]
    float* lastout = out + (size_t)B_Q * D_F;             // [N]

    char* ws = (char*)d_ws;
    _Float16* kh  = (_Float16*)ws;                                        // 32 MB
    _Float16* qh  = (_Float16*)(ws + (size_t)N_K * D_F * 2);              // 1 MB
    float*    cvb = (float*)(ws + (size_t)N_K * D_F * 2 + (size_t)B_Q * D_F * 2);
    int*      cib = (int*)((char*)cvb + (size_t)B_Q * NSPLIT * TOPK * 4);
    int*  maxslot = (int*)((char*)cib + (size_t)B_Q * NSPLIT * TOPK * 4);

    init_max_kernel<<<1, 32, 0, stream>>>(maxslot);
    prep_lastused_kernel<<<N_K / 256, 256, 0, stream>>>(lu, lastout, maxslot);
    q_normalize_kernel<<<B_Q / 8, 256, 0, stream>>>(q_in, qh);
    k_convert_kernel<<<(N_K * (D_F / 4)) / 256, 256, 0, stream>>>(K_in, kh);

    dim3 grid(B_Q / MTILE, NSPLIT);
    topk_gemm_kernel<<<grid, 256, WAVES * WAVE_LDS, stream>>>(qh, kh, cvb, cib);

    merge_softmax_out_kernel<<<B_Q / 8, 256, 0, stream>>>(cvb, cib, V_in, maxslot, out, lastout);
}
